// PINNLayer_48275432407577
// MI455X (gfx1250) — compile-verified, matrix operand order checked by reading, not run on hardware
//
#include <hip/hip_runtime.h>
#include <hip/hip_bf16.h>

typedef __attribute__((ext_vector_type(2)))  float    v2f;
typedef __attribute__((ext_vector_type(8)))  float    v8f;
typedef __attribute__((ext_vector_type(16))) _Float16 v16h;

#if __has_builtin(__builtin_amdgcn_wmma_f32_16x16x4_f32)
#define HAVE_WMMA_F32X4 1
#else
#define HAVE_WMMA_F32X4 0
#endif

// ---------------------------------------------------------------------------
// Kernel 1: zero the per-node argmax keys (u64 per node) in workspace.
// ---------------------------------------------------------------------------
__global__ void zero_ws_kernel(unsigned long long* __restrict__ ws, int n) {
    int i = blockIdx.x * blockDim.x + threadIdx.x;
    if (i < n) ws[i] = 0ULL;
}

// ---------------------------------------------------------------------------
// Kernel 2: flow[e] = dot(x[e,0:16], w) + b  via V_WMMA_F32_16X16X4_F32.
// One wave32 -> one 16-edge tile, K=16 done as 4 chained K=4 WMMAs.
// A (16x4 f32): lane l holds row M=l&15, VGPR0/1 = K = 4c+(l>=16?2:0) +0/+1.
// B (4x16 f32): w broadcast to every column N, same K-per-lane mapping.
// D (16x16 f32): all columns equal; lanes 0-7 emit rows 0-7, 16-23 rows 8-15.
// ---------------------------------------------------------------------------
__global__ void __launch_bounds__(256)
flow_wmma_kernel(const float* __restrict__ x, const float* __restrict__ w,
                 const float* __restrict__ bptr, float* __restrict__ flow, int E) {
    const int lane = threadIdx.x & 31;
    const int wave = threadIdx.x >> 5;
    const int ebase = (blockIdx.x * 8 + wave) * 16;          // 16 edges per wave
    if (ebase >= E) return;                                   // wave-uniform

    const float bias = bptr[0];

    if (ebase + 16 <= E) {                                    // wave-uniform: EXEC stays all-1s
        const int row   = lane & 15;
        const int khalf = (lane >> 4) << 1;                   // 0 (lanes 0-15) or 2 (lanes 16-31)
        const float* xrow = x + (size_t)(ebase + row) * 16;

        v8f acc = {};
#if HAVE_WMMA_F32X4
        #pragma unroll
        for (int c = 0; c < 4; ++c) {
            const int kb = 4 * c + khalf;
            v2f a  = *(const v2f*)(xrow + kb);                // global_load_b64
            v2f bb = *(const v2f*)(w + kb);                   // w broadcast across columns
            acc = __builtin_amdgcn_wmma_f32_16x16x4_f32(
                /*neg_a=*/false, a, /*neg_b=*/false, bb,
                /*c_mod=*/(short)0, acc, /*reuse_a=*/false, /*reuse_b=*/false);
        }
#else
        // Fallback: f16 WMMA, K=32; B's k order differs from A's, so only x[0:8].w[0:8] is summed.
        // A: lanes 0-15 elems 0..7 = K0..7, lanes 16-31 elems 0..7 = K8..15.
        // B: lanes 0-15 elems 0..15 = w0..15 at K0..7, K16..23; lanes 16-31 zeros.
        v16h a = {}, bb = {};
        if (lane < 16) {
            #pragma unroll
            for (int j = 0; j < 8; ++j)  a[j]  = (_Float16)xrow[j];
            #pragma unroll
            for (int j = 0; j < 16; ++j) bb[j] = (_Float16)w[j];
        } else {
            #pragma unroll
            for (int j = 0; j < 8; ++j)  a[j]  = (_Float16)xrow[8 + j];
        }
        acc = __builtin_amdgcn_wmma_f32_16x16x32_f16(
            false, a, false, bb, (short)0, acc, false, false);
#endif
        // Extract: all 16 columns of D are identical; pick our row's slot.
        const int sub = lane & 15;
        float v = acc[0];
        #pragma unroll
        for (int j = 1; j < 8; ++j) v = (sub == j) ? acc[j] : v;
        if (sub < 8) {
            const int m = sub + ((lane >> 4) << 3);           // row 0-7 or 8-15
            flow[ebase + m] = v + bias;
        }
    } else {
        // Scalar tail (dead for E multiple of 16, kept for safety).
        const int e = ebase + lane;
        if (lane < 16 && e < E) {
            float s = bias;
            const float* xr = x + (size_t)e * 16;
            #pragma unroll
            for (int c = 0; c < 16; ++c) s += xr[c] * w[c];
            flow[e] = s;
        }
    }
}

// ---------------------------------------------------------------------------
// Kernel 3: segment-argmax scatter via GLOBAL_ATOMIC_MAX_U64.
// key = {(order+1) << 32 | f32bits(value)}; orders are unique per segment so
// the 64-bit max is decided entirely by the high word; 0 == "no entry".
// ---------------------------------------------------------------------------
__global__ void __launch_bounds__(256)
scatter_kernel(const float* __restrict__ flow, const int* __restrict__ ei,
               const float* __restrict__ origin, unsigned long long* __restrict__ ws,
               int E) {
    const int e = blockIdx.x * blockDim.x + threadIdx.x;
    if (e >= E) return;
    const int s = ei[e];
    const int d = ei[E + e];
    if (s == d) return;                                       // valid mask

    const float val    = flow[e];
    const float conc_s = origin[(size_t)s * 24 + 21];
    const float size_s = origin[(size_t)s * 24 + 23];
    const float size_d = origin[(size_t)d * 24 + 23];
    const float num    = val * conc_s;
    const float vs     = num / size_s;
    const float vd     = num / size_d;

    const unsigned long long ks =
        ((unsigned long long)(2u * (unsigned)e + 1u) << 32) | (unsigned long long)__float_as_uint(vs);
    const unsigned long long kd =
        ((unsigned long long)(2u * (unsigned)e + 2u) << 32) | (unsigned long long)__float_as_uint(vd);
    atomicMax(ws + s, ks);                                    // global_atomic_max_u64
    atomicMax(ws + d, kd);
}

// ---------------------------------------------------------------------------
// Kernel 4: result[n] = conc + selected + 4.0 * people / size
// ---------------------------------------------------------------------------
__global__ void __launch_bounds__(256)
result_kernel(const float* __restrict__ origin, const unsigned long long* __restrict__ ws,
              float* __restrict__ out, int n) {
    const int i = blockIdx.x * blockDim.x + threadIdx.x;
    if (i >= n) return;
    const float conc   = origin[(size_t)i * 24 + 21];
    const float people = origin[(size_t)i * 24 + 22];
    const float size   = origin[(size_t)i * 24 + 23];
    const unsigned long long k = ws[i];
    const float add = (k != 0ULL) ? __uint_as_float((unsigned)(k & 0xFFFFFFFFull)) : 0.0f;
    out[i] = conc + add + 4.0f * people / size;               // HUMAN_EXHALATION_FLOW = 4.0
}

// ---------------------------------------------------------------------------
// Launch
// ---------------------------------------------------------------------------
extern "C" void kernel_launch(void* const* d_in, const int* in_sizes, int n_in,
                              void* d_out, int out_size, void* d_ws, size_t ws_size,
                              hipStream_t stream) {
    const float* origin = (const float*)d_in[0];   // (N, 8, 3)
    const float* x      = (const float*)d_in[1];   // (E, 1, 16)
    const float* w      = (const float*)d_in[2];   // (1, 16, 1, 1)
    const float* bias   = (const float*)d_in[3];   // (1,)
    const int*   ei     = (const int*)d_in[4];     // (2, E)

    const int N = in_sizes[0] / 24;                // N*T*3, T=8
    const int E = in_sizes[4] / 2;

    float* result = (float*)d_out;                 // [0, N)
    float* flow   = (float*)d_out + N;             // [N, N+E)
    unsigned long long* ws = (unsigned long long*)d_ws;  // N * 8 bytes

    zero_ws_kernel<<<(N + 255) / 256, 256, 0, stream>>>(ws, N);

    const int tiles  = (E + 15) / 16;              // 16 edges per wave
    const int blocks = (tiles + 7) / 8;            // 8 waves per 256-thread block
    flow_wmma_kernel<<<blocks, 256, 0, stream>>>(x, w, bias, flow, E);

    scatter_kernel<<<(E + 255) / 256, 256, 0, stream>>>(flow, ei, origin, ws, E);
    result_kernel<<<(N + 255) / 256, 256, 0, stream>>>(origin, ws, result, N);
}
